// VisualAttn_17695265259923
// MI455X (gfx1250) — compile-verified, matrix operand order checked by reading, not run on hardware
//
#include <hip/hip_runtime.h>
#include <hip/hip_bf16.h>
#include <cstddef>

// ---------------------------------------------------------------------------
// Types for CDNA5 WMMA (wave32): v_wmma_f32_16x16x32_bf16
//   A: 16x32 bf16  -> 16 bf16 / lane  (v16bf)
//   B: 32x16 bf16  -> 16 bf16 / lane  (v16bf)
//   C/D: 16x16 f32 ->  8 f32  / lane  (v8f)
// ---------------------------------------------------------------------------
typedef __attribute__((ext_vector_type(16))) __bf16 v16bf;
typedef __attribute__((ext_vector_type(8)))  __bf16 v8bf;
typedef __attribute__((ext_vector_type(8)))  float  v8f;

#define B_  16
#define F_  64
#define E_  16
#define HW_ 4096   // 64*64
#define KTOT 576   // F*9
#define KSTEPS 18  // 576 / 32

// ---------------------------------------------------------------------------
// Kernel P1: state NCHW f32 -> NHWC bf16   (GEMM-K = channels contiguous)
// idx = ((b*4096 + p)*64 + c)
// ---------------------------------------------------------------------------
__global__ __launch_bounds__(256) void to_nhwc_bf16(const float* __restrict__ in,
                                                    __bf16* __restrict__ out) {
  int idx = blockIdx.x * 256 + threadIdx.x;
  int c = idx & 63;
  int p = (idx >> 6) & 4095;
  int b = idx >> 18;
  out[idx] = (__bf16)in[((size_t)(b * 64 + c)) * 4096 + p];
}

// ---------------------------------------------------------------------------
// Kernel P2: pack conv weights (O,64,3,3) f32 into per-lane B-fragment image:
//   out[((t*18 + s)*32 + lane)*16 + j]  (bf16)
// K ordering: k = tap*64 + c  (tap = 3x3 position, c = input channel).
// 16-bit B fragment layout (mirror of A): lane<16 -> N=lane, K = kb0+{0..7},
// kb0+16+{0..7} with kb0 = (lane>>4)*8.
// ---------------------------------------------------------------------------
__global__ __launch_bounds__(256) void pack_wB(const float* __restrict__ w,
                                               __bf16* __restrict__ out,
                                               int ntiles) {
  int idx = blockIdx.x * 256 + threadIdx.x;
  int total = ntiles * KSTEPS * 32 * 16;
  if (idx >= total) return;
  int j = idx & 15;
  int l = (idx >> 4) & 31;
  int s = (idx >> 9) % KSTEPS;
  int t = idx / (KSTEPS * 512);
  int o   = t * 16 + (l & 15);
  int kb0 = (l >> 4) * 8;
  int kk  = (j < 8) ? (kb0 + j) : (kb0 + 16 + (j - 8));
  int k   = s * 32 + kk;
  int c   = k & 63;
  int tap = k >> 6;
  int ky = tap / 3, kx = tap % 3;
  out[idx] = (__bf16)w[(((size_t)o * 64 + c) * 3 + ky) * 3 + kx];
}

// ---------------------------------------------------------------------------
// Conv3x3 'same' as implicit GEMM on WMMA bf16.
// grid = B*32 workgroups, 256 threads (8 waves). Each wave: 16 contiguous
// pixels (M) x NTILES*16 output channels. A fragments loaded directly from
// NHWC bf16 global memory (two contiguous b128 per lane), halo -> zeros.
// ---------------------------------------------------------------------------
template <int NTILES, bool SIGMOID>
__global__ __launch_bounds__(256) void conv3x3_wmma(
    const __bf16* __restrict__ src,   // (B,64,64,64ch) bf16
    const __bf16* __restrict__ wB,    // packed B fragments
    const float*  __restrict__ bias,  // (NTILES*16)
    __bf16* __restrict__ out_bf,      // NHWC bf16 (relu path)
    float*  __restrict__ out_f32) {   // NCHW f32 (sigmoid path)
  const int tid  = threadIdx.x;
  const int wave = tid >> 5;
  const int lane = tid & 31;
  const int b    = blockIdx.x >> 5;   // 32 WGs per batch
  const int wgp  = blockIdx.x & 31;
  const int pbase = wgp * 128 + wave * 16;
  const int m  = lane & 15;           // pixel row within the 16x16 tile
  const int p  = pbase + m;
  const int y  = p >> 6, x = p & 63;
  const int kb = (lane >> 4) * 8;     // K sub-offset per A layout

  v8f acc[NTILES];
#pragma unroll
  for (int t = 0; t < NTILES; ++t)
#pragma unroll
    for (int i = 0; i < 8; ++i) acc[t][i] = 0.0f;

#pragma unroll
  for (int s = 0; s < KSTEPS; ++s) {
    const int tap = s >> 1;
    const int c0  = (s & 1) * 32;
    const int py  = y + tap / 3 - 1;
    const int px  = x + tap % 3 - 1;

    v16bf a;
#pragma unroll
    for (int i = 0; i < 16; ++i) a[i] = (__bf16)0.0f;
    if ((unsigned)py < 64u && (unsigned)px < 64u) {
      const __bf16* ap =
          src + (((size_t)(b * 64 + py) * 64 + px) * 64 + c0 + kb);
      const v8bf lo = *(const v8bf*)(ap);       // K = kb .. kb+7
      const v8bf hi = *(const v8bf*)(ap + 16);  // K = kb+16 .. kb+23
#pragma unroll
      for (int i = 0; i < 8; ++i) { a[i] = lo[i]; a[i + 8] = hi[i]; }
    }

#pragma unroll
    for (int t = 0; t < NTILES; ++t) {
      const v16bf bf =
          *(const v16bf*)(wB + (((size_t)t * KSTEPS + s) * 32 + lane) * 16);
      acc[t] = __builtin_amdgcn_wmma_f32_16x16x32_bf16(
          false, a, false, bf, (short)0, acc[t], false, false);
    }
  }

  // Epilogue: bias + nonlinearity + store.
#pragma unroll
  for (int t = 0; t < NTILES; ++t) {
    const int ch = t * 16 + (lane & 15);
    const float bv = bias[ch];
#pragma unroll
    for (int r = 0; r < 8; ++r) {
      const int mm  = r + (lane >> 4) * 8;  // C-layout row
      const int pix = pbase + mm;
      float v = acc[t][r] + bv;
      if (SIGMOID) {
        v = 1.0f / (1.0f + __expf(-v));
        out_f32[((size_t)(b * NTILES * 16) + ch) * 4096 + pix] = v;  // NCHW
      } else {
        v = v > 0.0f ? v : 0.0f;
        out_bf[((size_t)b * 4096 + pix) * 64 + ch] = (__bf16)v;      // NHWC
      }
    }
  }
}

// ---------------------------------------------------------------------------
// Fused t = c3_w @ state (pointwise over channels) + ReLU reduction.
// out_partial[b,chunk,e,o] = sum_{p in chunk} relu(attn[b,e,p]*t[b,o,p]+c3_b[o])
// grid = B*8, 256 threads. LDS: c3_w(16K) + S(16K) + T(16K) + attn(4K) = 52KB.
// ---------------------------------------------------------------------------
__global__ __launch_bounds__(256) void fuse_reduce(
    const float* __restrict__ state,  // (B,64,4096) NCHW f32
    const float* __restrict__ attn,   // (B,16,4096) f32 (in d_out)
    const float* __restrict__ c3w,    // (64,64)
    const float* __restrict__ c3b,    // (64)
    float* __restrict__ partial) {    // (B,8,16,64)
  __shared__ float sW[64 * 64];
  __shared__ float sS[64 * 64];
  __shared__ float sT[64 * 64];
  __shared__ float sA[16 * 64];
  const int tid   = threadIdx.x;
  const int b     = blockIdx.x >> 3;
  const int chunk = blockIdx.x & 7;

  for (int i = tid; i < 4096; i += 256) sW[i] = c3w[i];

  float acc[4] = {0.f, 0.f, 0.f, 0.f};

  const int pstart = chunk * 512;
  for (int pb = pstart; pb < pstart + 512; pb += 64) {
    __syncthreads();  // previous phase2 done before reloading sS/sA
    for (int i = tid; i < 4096; i += 256) {
      int c = i >> 6, pp = i & 63;
      sS[i] = state[(size_t)(b * 64 + c) * 4096 + pb + pp];
    }
    for (int i = tid; i < 1024; i += 256) {
      int e = i >> 6, pp = i & 63;
      sA[i] = attn[(size_t)(b * 16 + e) * 4096 + pb + pp];
    }
    __syncthreads();
    // phase 1: T[p][o] = dot(c3w[o,:], S[:,p])  (16 entries per thread)
    {
      const int o = tid & 63;
      const int pb2 = (tid >> 6) * 16;
      for (int ii = 0; ii < 16; ++ii) {
        const int pp = pb2 + ii;
        float s = 0.f;
#pragma unroll 8
        for (int c = 0; c < 64; ++c) s += sW[o * 64 + c] * sS[c * 64 + pp];
        sT[pp * 64 + o] = s;
      }
    }
    __syncthreads();
    // phase 2: relu-reduce over the 64 pixels; 4 (e,o) pairs per thread
#pragma unroll
    for (int j = 0; j < 4; ++j) {
      const int q = tid + 256 * j;
      const int e = q >> 6, o = q & 63;
      const float bv = c3b[o];
      float a = 0.f;
#pragma unroll 8
      for (int pp = 0; pp < 64; ++pp) {
        float v = sA[e * 64 + pp] * sT[pp * 64 + o] + bv;
        a += v > 0.f ? v : 0.f;
      }
      acc[j] += a;
    }
  }
#pragma unroll
  for (int j = 0; j < 4; ++j) {
    const int q = tid + 256 * j;
    const int e = q >> 6, o = q & 63;
    partial[(((size_t)b * 8 + chunk) * 16 + e) * 64 + o] = acc[j];
  }
}

// ---------------------------------------------------------------------------
// Finalize: out[b,e,o] = (sum_chunks + 260*relu(c3_b[o])) / 4356
// (260 = 66*66 - 64*64 zero-pad border pixels of the (1,1,1) conv3d)
// ---------------------------------------------------------------------------
__global__ __launch_bounds__(256) void finalize(const float* __restrict__ partial,
                                                const float* __restrict__ c3b,
                                                float* __restrict__ out) {
  int idx = blockIdx.x * 256 + threadIdx.x;  // 16384 = B*E*F
  int o = idx & 63;
  int e = (idx >> 6) & 15;
  int b = idx >> 10;
  float s = 0.f;
#pragma unroll
  for (int ch = 0; ch < 8; ++ch)
    s += partial[(((size_t)b * 8 + ch) * 16 + e) * 64 + o];
  float bb = c3b[o];
  bb = bb > 0.f ? bb : 0.f;
  out[idx] = (s + 260.0f * bb) * (1.0f / 4356.0f);
}

// ---------------------------------------------------------------------------
extern "C" void kernel_launch(void* const* d_in, const int* in_sizes, int n_in,
                              void* d_out, int out_size, void* d_ws,
                              size_t ws_size, hipStream_t stream) {
  (void)in_sizes; (void)n_in; (void)out_size; (void)ws_size;
  const float* state  = (const float*)d_in[0];  // (16,64,64,64)
  const float* pre_w  = (const float*)d_in[1];  // (64,64,3,3)
  const float* pre_b  = (const float*)d_in[2];  // (64)
  const float* attn_w = (const float*)d_in[3];  // (16,64,3,3)
  const float* attn_b = (const float*)d_in[4];  // (16)
  const float* c3_w   = (const float*)d_in[5];  // (64,64)
  const float* c3_b   = (const float*)d_in[6];  // (64)

  float* out  = (float*)d_out;      // (16,16,64) = 16384 floats
  float* attn = out + 16384;        // (16,16,64,64) = 1048576 floats

  char* ws = (char*)d_ws;
  __bf16* state_bf = (__bf16*)(ws);                 // 8,388,608 B
  __bf16* x_bf     = (__bf16*)(ws + 8388608);       // 8,388,608 B
  __bf16* wB1      = (__bf16*)(ws + 16777216);      //    73,728 B
  __bf16* wB2      = (__bf16*)(ws + 16850944);      //    18,432 B
  float*  partial  = (float*) (ws + 16869376);      //   524,288 B

  // 1) layout transform + precision cast
  to_nhwc_bf16<<<16384, 256, 0, stream>>>(state, state_bf);
  // 2) pack weight B-fragments
  pack_wB<<<144, 256, 0, stream>>>(pre_w, wB1, 4);
  pack_wB<<<36, 256, 0, stream>>>(attn_w, wB2, 1);
  // 3) conv1 + relu (WMMA bf16)
  conv3x3_wmma<4, false><<<512, 256, 0, stream>>>(state_bf, wB1, pre_b, x_bf,
                                                  nullptr);
  // 4) conv2 + sigmoid -> attn output (WMMA bf16)
  conv3x3_wmma<1, true><<<512, 256, 0, stream>>>(x_bf, wB2, attn_b, nullptr,
                                                 attn);
  // 5) fused pointwise GEMM + ReLU reduction (deterministic partials)
  fuse_reduce<<<128, 256, 0, stream>>>(state, attn, c3_w, c3_b, partial);
  // 6) finalize mean (+ zero-pad border closed form)
  finalize<<<64, 256, 0, stream>>>(partial, c3_b, out);
}
